// MoleculeGCN_43430709297638
// MI455X (gfx1250) — compile-verified, matrix operand order checked by reading, not run on hardware
//
#include <hip/hip_runtime.h>
#include <hip/hip_bf16.h>

typedef __attribute__((ext_vector_type(16))) __bf16 v16bf;
typedef __attribute__((ext_vector_type(8)))  float  v8f;

#define N_NODES 100000
#define N_EDGES 400000
#define N_GRAPH 4096
#define EMB     128

// ---------------- helpers ----------------
static __device__ __forceinline__ float sp(float x) {
  // numerically stable softplus: log1p(exp(-|x|)) + max(x,0)
  return log1pf(expf(-fabsf(x))) + fmaxf(x, 0.f);
}
static __device__ __forceinline__ float bf2f(unsigned short h) {
  return __uint_as_float(((unsigned int)h) << 16);
}
static __device__ __forceinline__ unsigned short f2bf(float f) {
  unsigned int u = __float_as_uint(f);
  return (unsigned short)((u + 0x7FFFu + ((u >> 16) & 1u)) >> 16);  // RNE
}
static __device__ __forceinline__ void atomicMaxF(float* a, float v) {
  // order-isomorphic int trick; correct with -inf init, mixed signs
  if (v >= 0.f) atomicMax((int*)a, __float_as_int(v));
  else          atomicMin((unsigned int*)a, __float_as_uint(v));
}

// ---------------- small kernels ----------------
__global__ void k_embed(const int* __restrict__ ids, const float* __restrict__ tab,
                        float* __restrict__ of, unsigned short* __restrict__ ob) {
  int i = blockIdx.x * 256 + threadIdx.x;            // grid sized exactly rows*128/256
  int r = i >> 7, c = i & 127;
  float v = sp(tab[(size_t)ids[r] * EMB + c]);
  of[i] = v;
  ob[i] = f2bf(v);
}

__global__ void k_fill_neginf(float* __restrict__ p) {
  int i = blockIdx.x * 256 + threadIdx.x;
  p[i] = __uint_as_float(0xFF800000u);               // -inf (segment_max identity)
}

__global__ void k_count(const int* __restrict__ dst, int* __restrict__ cnt) {
  int i = blockIdx.x * 256 + threadIdx.x;
  if (i < N_EDGES) atomicAdd(&cnt[dst[i]], 1);
}

// pack weights (fp32 [256,128], rows 0..127 from W0, 128..255 from W1) into
// per-lane bf16 WMMA B-fragments: idx = ((kt*8+ct)*32 + lane)*8 + v
__global__ void k_pack(const float* __restrict__ W0, const float* __restrict__ W1,
                       unsigned int* __restrict__ Wp) {
  int idx = blockIdx.x * 256 + threadIdx.x;          // 16384 total
  int v  = idx & 7;
  int ln = (idx >> 3) & 31;
  int ct = (idx >> 8) & 7;
  int kt = (idx >> 11) & 7;
  int col = ct * 16 + (ln & 15);
  int hi  = ln >> 4;
  int k0 = kt * 32 + ((v & 4) << 2) + 2 * (v & 3) + 8 * hi;
  float f0 = (k0 < 128) ? W0[(size_t)k0 * EMB + col] : W1[(size_t)(k0 - 128) * EMB + col];
  int k1 = k0 + 1;
  float f1 = (k1 < 128) ? W0[(size_t)k1 * EMB + col] : W1[(size_t)(k1 - 128) * EMB + col];
  Wp[idx] = (unsigned int)f2bf(f0) | ((unsigned int)f2bf(f1) << 16);
}

// ---------------- fused edge GEMM + scatter (WMMA) ----------------
// MODE 0: EdgeConv  A = [hb[dst] | hb[src]-hb[dst]] @ Wb + b_bond  -> atomicMax into hb_f[dst]
// MODE 1: GenConv   A = [ha[src] | hb[e]]           @ Wme + b_m+b_e -> atomicAdd into ha_f[dst]
template <int MODE>
__global__ __launch_bounds__(256) void k_edge_gemm(
    const int* __restrict__ src, const int* __restrict__ dst,
    const unsigned short* __restrict__ ha_bf, const unsigned short* __restrict__ hb_bf,
    const unsigned int* __restrict__ Wp,
    const float* __restrict__ bias1, const float* __restrict__ bias2,
    float* __restrict__ outf) {
  __shared__ unsigned short A[16 * 256];   // 16 edge-rows x K=256, bf16
  __shared__ int drow[16];
  const int t = threadIdx.x;
  const int ebase = blockIdx.x * 16;

  // ---- stage A tile: 16 threads per row, 8 cols-per-half per thread (uint4) ----
  {
    const int r  = t >> 4;
    const int c0 = (t & 15) * 8;
    const int e  = ebase + r;
    if ((t & 15) == 0) drow[r] = dst[e];
    if (MODE == 0) {
      const unsigned short* ri = hb_bf + (size_t)dst[e] * EMB + c0;
      const unsigned short* rj = hb_bf + (size_t)src[e] * EMB + c0;
      uint4 vi = *(const uint4*)ri;
      uint4 vj = *(const uint4*)rj;
      *(uint4*)&A[r * 256 + c0] = vi;
      const unsigned int* pi = (const unsigned int*)&vi;
      const unsigned int* pj = (const unsigned int*)&vj;
      unsigned int di[4];
#pragma unroll
      for (int q = 0; q < 4; ++q) {
        float a0 = bf2f((unsigned short)(pj[q] & 0xFFFFu)) - bf2f((unsigned short)(pi[q] & 0xFFFFu));
        float a1 = bf2f((unsigned short)(pj[q] >> 16))     - bf2f((unsigned short)(pi[q] >> 16));
        di[q] = (unsigned int)f2bf(a0) | ((unsigned int)f2bf(a1) << 16);
      }
      *(uint4*)&A[r * 256 + 128 + c0] = *(uint4*)di;
    } else {
      const unsigned short* ri = ha_bf + (size_t)src[e] * EMB + c0;
      const unsigned short* rj = hb_bf + (size_t)e * EMB + c0;
      *(uint4*)&A[r * 256 + c0]       = *(const uint4*)ri;
      *(uint4*)&A[r * 256 + 128 + c0] = *(const uint4*)rj;
    }
  }
  __syncthreads();

  // ---- WMMA: each wave owns one 16-wide column tile; K = 256 in 8 steps ----
  const int lane = t & 31;
  const int ct   = t >> 5;                 // wave id == column tile (8 waves)
  const int M    = lane & 15;
  const int hi   = lane >> 4;
  union { unsigned int u[8]; v16bf v; } a, b;
  union { float f[8]; v8f v; } c;
#pragma unroll
  for (int q = 0; q < 8; ++q) c.f[q] = 0.f;

#pragma unroll
  for (int kt = 0; kt < 8; ++kt) {
#pragma unroll
    for (int q = 0; q < 8; ++q) {
      int k = kt * 32 + ((q & 4) << 2) + 2 * (q & 3) + 8 * hi;   // 16-bit A layout
      a.u[q] = *(const unsigned int*)&A[M * 256 + k];
    }
    const uint4* bp = (const uint4*)(Wp + (((kt * 8 + ct) * 32 + lane) << 3));
    uint4 b0 = bp[0], b1 = bp[1];
    b.u[0] = b0.x; b.u[1] = b0.y; b.u[2] = b0.z; b.u[3] = b0.w;
    b.u[4] = b1.x; b.u[5] = b1.y; b.u[6] = b1.z; b.u[7] = b1.w;
    c.v = __builtin_amdgcn_wmma_f32_16x16x32_bf16(false, a.v, false, b.v,
                                                  (short)0, c.v, false, false);
  }

  // ---- epilogue: bias + scatter (segment max / segment add) ----
  const int col = ct * 16 + M;
  float bias = bias1[col] + ((MODE == 1) ? bias2[col] : 0.f);
#pragma unroll
  for (int q = 0; q < 8; ++q) {
    int row = q + 8 * hi;                  // f32 C/D layout: M = vgpr + 8*hi
    int d = drow[row];
    float val = c.f[q] + bias;
    if (MODE == 0) atomicMaxF(&outf[(size_t)d * EMB + col], val);
    else           atomicAdd(&outf[(size_t)d * EMB + col], val);
  }
}

// ---------------- finalize / pool / readout ----------------
__global__ void k_fin_hb(const int* __restrict__ cnt, float* __restrict__ hf,
                         unsigned short* __restrict__ hb) {
  int i = blockIdx.x * 256 + threadIdx.x;
  int r = i >> 7;
  float x = (cnt[r] > 0) ? hf[i] : 0.f;    // where(has_nb, agg, 0)
  float h = sp(x);
  hf[i] = h;
  hb[i] = f2bf(h);
}

__global__ void k_fin_ha(float* __restrict__ hf, unsigned short* __restrict__ hb) {
  int i = blockIdx.x * 256 + threadIdx.x;
  float h = sp(hf[i]);                     // hf already holds segsum + old h_a
  hf[i] = h;
  hb[i] = f2bf(h);
}

__global__ void k_pool(const int* __restrict__ batch, const float* __restrict__ ha,
                       float* __restrict__ pooled) {
  int i = blockIdx.x * 256 + threadIdx.x;
  int r = i >> 7, c = i & 127;
  atomicAdd(&pooled[(size_t)batch[r] * EMB + c], ha[i]);
}

__global__ __launch_bounds__(128) void k_readout(
    const float* __restrict__ pooled,
    const float* __restrict__ W1, const float* __restrict__ b1,
    const float* __restrict__ W2, const float* __restrict__ b2,
    const float* __restrict__ W3, const float* __restrict__ b3,
    float* __restrict__ out) {
  __shared__ float p[128], h1[64], h2[64];
  const int t = threadIdx.x, g = blockIdx.x;
  p[t] = pooled[(size_t)g * EMB + t];
  __syncthreads();
  if (t < 64) {
    float s = b1[t];
#pragma unroll 4
    for (int k = 0; k < 128; ++k) s += p[k] * W1[k * 64 + t];
    h1[t] = sp(s);
  }
  __syncthreads();
  if (t < 64) {
    float s = b2[t];
#pragma unroll 4
    for (int k = 0; k < 64; ++k) s += h1[k] * W2[k * 64 + t];
    h2[t] = sp(s);
  }
  __syncthreads();
  if (t == 0) {
    float s = b3[0];
    for (int k = 0; k < 64; ++k) s += h2[k] * W3[k];
    out[g] = s;
  }
}

// ---------------- driver ----------------
extern "C" void kernel_launch(void* const* d_in, const int* in_sizes, int n_in,
                              void* d_out, int out_size, void* d_ws, size_t ws_size,
                              hipStream_t stream) {
  (void)in_sizes; (void)n_in; (void)out_size; (void)ws_size;
  const int*   x_ids    = (const int*)d_in[0];
  const int*   eattr    = (const int*)d_in[1];
  const int*   eidx     = (const int*)d_in[2];
  const int*   batch    = (const int*)d_in[3];
  const float* emb_atom = (const float*)d_in[4];
  const float* emb_bond = (const float*)d_in[5];
  const float* W_bond   = (const float*)d_in[6];
  const float* b_bond   = (const float*)d_in[7];
  const float* W_msg    = (const float*)d_in[8];
  const float* b_msg    = (const float*)d_in[9];
  const float* W_edge   = (const float*)d_in[10];
  const float* b_edge   = (const float*)d_in[11];
  const float* W_r1     = (const float*)d_in[12];
  const float* b_r1     = (const float*)d_in[13];
  const float* W_r2     = (const float*)d_in[14];
  const float* b_r2     = (const float*)d_in[15];
  const float* W_r3     = (const float*)d_in[16];
  const float* b_r3     = (const float*)d_in[17];

  const int* src = eidx;                 // edge_index[0]
  const int* dst = eidx + N_EDGES;       // edge_index[1]

  // d_out = [out(G) | h_a(N*128) | h_b(E*128)]; use h_a/h_b regions as fp32 state
  float* out  = (float*)d_out;
  float* ha_f = out + N_GRAPH;
  float* hb_f = ha_f + (size_t)N_NODES * EMB;

  char* ws = (char*)d_ws;
  unsigned short* ha_bf = (unsigned short*)ws;  ws += (size_t)N_NODES * EMB * 2;  // 25.6 MB
  unsigned short* hb_bf = (unsigned short*)ws;  ws += (size_t)N_EDGES * EMB * 2;  // 102.4 MB
  int*            cnt   = (int*)ws;             ws += (size_t)N_EDGES * 4;        // 1.6 MB
  unsigned int*   Wb_p  = (unsigned int*)ws;    ws += 16384 * 4;                  // 64 KB
  unsigned int*   Wme_p = (unsigned int*)ws;    ws += 16384 * 4;                  // 64 KB
  float*          pooled= (float*)ws;           ws += (size_t)N_GRAPH * EMB * 4;  // 2 MB

  // weight packing (bf16 WMMA B fragments)
  k_pack<<<64, 256, 0, stream>>>(W_bond, W_bond + 128 * EMB, Wb_p);
  k_pack<<<64, 256, 0, stream>>>(W_msg, W_edge, Wme_p);

  // embeddings + softplus (fp32 state in d_out, bf16 mirror in ws)
  k_embed<<<(N_NODES * EMB) / 256, 256, 0, stream>>>(x_ids, emb_atom, ha_f, ha_bf);
  k_embed<<<(N_EDGES * EMB) / 256, 256, 0, stream>>>(eattr, emb_bond, hb_f, hb_bf);

  // incoming-edge counts over dst (num_segments = E, dst < N)
  hipMemsetAsync(cnt, 0, (size_t)N_EDGES * 4, stream);
  k_count<<<(N_EDGES + 255) / 256, 256, 0, stream>>>(dst, cnt);

  for (int step = 0; step < 2; ++step) {
    // EdgeConv: agg lives in hb_f (fp32), gathers use stale hb_bf
    k_fill_neginf<<<(N_EDGES * EMB) / 256, 256, 0, stream>>>(hb_f);
    k_edge_gemm<0><<<N_EDGES / 16, 256, 0, stream>>>(src, dst, ha_bf, hb_bf, Wb_p,
                                                     b_bond, b_bond, hb_f);
    k_fin_hb<<<(N_EDGES * EMB) / 256, 256, 0, stream>>>(cnt, hb_f, hb_bf);
    // GeneralConv: accumulate into ha_f (already holds old h_a = identity term)
    k_edge_gemm<1><<<N_EDGES / 16, 256, 0, stream>>>(src, dst, ha_bf, hb_bf, Wme_p,
                                                     b_msg, b_edge, ha_f);
    k_fin_ha<<<(N_NODES * EMB) / 256, 256, 0, stream>>>(ha_f, ha_bf);
  }

  // global_add_pool + readout MLP
  hipMemsetAsync(pooled, 0, (size_t)N_GRAPH * EMB * 4, stream);
  k_pool<<<(N_NODES * EMB) / 256, 256, 0, stream>>>(batch, ha_f, pooled);
  k_readout<<<N_GRAPH, 128, 0, stream>>>(pooled, W_r1, b_r1, W_r2, b_r2, W_r3, b_r3, out);
}
